// EncoderLayer_2250562863254
// MI455X (gfx1250) — compile-verified, matrix operand order checked by reading, not run on hardware
//
#include <hip/hip_runtime.h>
#include <hip/hip_bf16.h>

// ---------------------------------------------------------------------------
// Transformer encoder layer for MI455X (gfx1250), f16 WMMA with f32 accum.
// All matrix math goes through v_wmma_f32_16x16x32_f16 (wave32).
// Tile staging uses GLOBAL_LOAD_ASYNC_TO_LDS_B128 (ASYNCcnt) when the
// toolchain exposes it, with a synchronous uint4 fallback otherwise.
// ---------------------------------------------------------------------------

typedef __attribute__((ext_vector_type(16))) _Float16 v16h;
typedef __attribute__((ext_vector_type(8)))  float    v8f;
typedef __attribute__((ext_vector_type(4)))  int      v4i;

constexpr int MSEQ   = 1024;
constexpr int BATCH  = 8;
constexpr int DMODEL = 1024;
constexpr int NH     = 16;
constexpr int DHEAD  = 64;   // DMODEL / NH
constexpr int FFDIM  = 4096;
constexpr int ROWS   = MSEQ * BATCH;   // 8192 token rows
constexpr float NEG_INF_VAL = -1000.0f;

// --------------------------- async copy plumbing ---------------------------
#if defined(__has_builtin)
#  if __has_builtin(__builtin_amdgcn_global_load_async_to_lds_b128)
#    define HAVE_ASYNC_LDS 1
#  endif
#endif
#ifndef HAVE_ASYNC_LDS
#  define HAVE_ASYNC_LDS 0
#endif

#if HAVE_ASYNC_LDS && defined(__has_builtin) && __has_builtin(__builtin_amdgcn_s_wait_asynccnt)
#  define WAIT_ASYNC() __builtin_amdgcn_s_wait_asynccnt(0)
#elif HAVE_ASYNC_LDS
#  define WAIT_ASYNC() asm volatile("s_wait_asynccnt 0" ::: "memory")
#else
#  define WAIT_ASYNC() do {} while (0)
#endif

// Builtin prototype (from hipcc diagnostic): param pointee is int __vector(4),
// global (AS1) source first, LDS (AS3) destination second, imm offset, imm cpol.
typedef __attribute__((address_space(1))) v4i* gptr_v4i;
typedef __attribute__((address_space(3))) v4i* lptr_v4i;

// 16B global -> LDS copy; async (no VGPR round-trip, ASYNCcnt) when available.
__device__ __forceinline__ void copy_b128_to_lds(void* lds_dst, const void* gsrc) {
#if HAVE_ASYNC_LDS
    __builtin_amdgcn_global_load_async_to_lds_b128(
        (gptr_v4i)gsrc, (lptr_v4i)lds_dst, 0, 0);
#else
    *reinterpret_cast<uint4*>(lds_dst) = *reinterpret_cast<const uint4*>(gsrc);
#endif
}

// ---------------------------------------------------------------------------
// Fragment loader: CDNA5 16-bit A/B layout (CDNA5 ISA §7.12.2).
// Lane l (row = tileRow + l%16) holds, in v16h elements 0..7, k = koff..koff+7
// and in elements 8..15, k = 16+koff..16+koff+7, where koff = (l>=16) ? 8 : 0.
// Both halves are 16B-contiguous -> two ds_load_b128 per fragment.
// ---------------------------------------------------------------------------
__device__ __forceinline__ v16h frag_load(const _Float16* rowbase, int koff) {
    union { v16h h; struct { uint4 lo; uint4 hi; } u; } f;
    f.u.lo = *reinterpret_cast<const uint4*>(rowbase + koff);
    f.u.hi = *reinterpret_cast<const uint4*>(rowbase + koff + 16);
    return f.h;
}

// ---------------------------------------------------------------------------
// Weight transpose + fp32 -> f16 convert:  src (K x N) -> dst (N x K)
// ---------------------------------------------------------------------------
__global__ void __launch_bounds__(256)
transpose_convert_kernel(const float* __restrict__ src, _Float16* __restrict__ dst,
                         int K, int N) {
    __shared__ float tile[32][33];
    const int n  = blockIdx.x * 32 + threadIdx.x;
    const int k0 = blockIdx.y * 32;
    for (int i = threadIdx.y; i < 32; i += 8)
        tile[i][threadIdx.x] = src[(size_t)(k0 + i) * N + n];
    __syncthreads();
    const int kk = k0 + threadIdx.x;
    const int n0 = blockIdx.x * 32;
    for (int i = threadIdx.y; i < 32; i += 8)
        dst[(size_t)(n0 + i) * K + kk] = (_Float16)tile[threadIdx.x][i];
}

// ---------------------------------------------------------------------------
// LayerNorm over D=1024, one block per token row, writes f16.
// ---------------------------------------------------------------------------
__global__ void __launch_bounds__(256)
layernorm_kernel(const float* __restrict__ x, const float* __restrict__ g,
                 const float* __restrict__ bta, _Float16* __restrict__ out) {
    __shared__ float red0[8], red1[8];
    const int row = blockIdx.x;
    const int tid = threadIdx.x;
    const float4 xv = *reinterpret_cast<const float4*>(x + (size_t)row * DMODEL + tid * 4);
    float s  = xv.x + xv.y + xv.z + xv.w;
    float ss = xv.x * xv.x + xv.y * xv.y + xv.z * xv.z + xv.w * xv.w;
    #pragma unroll
    for (int off = 1; off < 32; off <<= 1) {
        s  += __shfl_xor(s,  off, 32);
        ss += __shfl_xor(ss, off, 32);
    }
    const int wid = tid >> 5, lane = tid & 31;
    if (lane == 0) { red0[wid] = s; red1[wid] = ss; }
    __syncthreads();
    s = 0.f; ss = 0.f;
    #pragma unroll
    for (int i = 0; i < 8; i++) { s += red0[i]; ss += red1[i]; }
    const float mu   = s * (1.0f / DMODEL);
    const float var  = ss * (1.0f / DMODEL) - mu * mu;
    const float rstd = rsqrtf(var + 1e-5f);
    const float4 gv = *reinterpret_cast<const float4*>(g   + tid * 4);
    const float4 bv = *reinterpret_cast<const float4*>(bta + tid * 4);
    _Float16* o = out + (size_t)row * DMODEL + tid * 4;
    o[0] = (_Float16)((xv.x - mu) * rstd * gv.x + bv.x);
    o[1] = (_Float16)((xv.y - mu) * rstd * gv.y + bv.y);
    o[2] = (_Float16)((xv.z - mu) * rstd * gv.z + bv.z);
    o[3] = (_Float16)((xv.w - mu) * rstd * gv.w + bv.w);
}

// ---------------------------------------------------------------------------
// Generic WMMA GEMM:  out = A(R x K, f16) @ W(K x N) + bias [, relu][, +resid]
// W supplied pre-transposed as Wt (N x K, f16) so A- and B-tiles load alike.
// Block tile 128x128, BK=32; 8 waves as 2(M) x 4(N); wave tile 64x32
// => 8 accumulators, 8 v_wmma_f32_16x16x32_f16 per k-step.
// Double-buffered LDS: tile k+1 prefetched (async when available) while
// tile k feeds the WMMAs; one s_wait_asynccnt + barrier per k-step.
// ---------------------------------------------------------------------------
constexpr int GT_M = 128, GT_N = 128, GT_K = 32;
constexpr int LDT  = 40;  // 32 + 8 pad f16 => 80B row stride (16B aligned)

template<bool RELU>
__global__ void __launch_bounds__(256)
gemm_f16_kernel(const _Float16* __restrict__ A, const _Float16* __restrict__ Wt,
                const float* __restrict__ bias, const float* __restrict__ resid,
                float* __restrict__ out32, _Float16* __restrict__ out16,
                int K, int N) {
    __shared__ __align__(16) _Float16 As[2][GT_M * LDT];
    __shared__ __align__(16) _Float16 Bs[2][GT_N * LDT];

    const int tid  = threadIdx.x;
    const int wid  = tid >> 5;
    const int lane = tid & 31;
    const int half = lane >> 4;
    const int ln   = lane & 15;
    const int koff = half ? 8 : 0;

    const int rowBase = blockIdx.x * GT_M;
    const int colBase = blockIdx.y * GT_N;
    const int wm = wid >> 2;   // 0..1 -> 64 rows
    const int wn = wid & 3;    // 0..3 -> 32 cols

    auto load_tiles = [&](int buf, int k0) {
        #pragma unroll
        for (int i = 0; i < 2; i++) {
            const int id  = tid + i * 256;      // 0..511
            const int r   = id >> 2;            // 0..127
            const int seg = (id & 3) << 3;      // 0,8,16,24
            copy_b128_to_lds(&As[buf][r * LDT + seg],
                             A  + (size_t)(rowBase + r) * K + k0 + seg);
            copy_b128_to_lds(&Bs[buf][r * LDT + seg],
                             Wt + (size_t)(colBase + r) * K + k0 + seg);
        }
    };

    v8f acc[4][2];
    #pragma unroll
    for (int mt = 0; mt < 4; mt++)
        #pragma unroll
        for (int nt = 0; nt < 2; nt++)
            #pragma unroll
            for (int r = 0; r < 8; r++) acc[mt][nt][r] = 0.0f;

    load_tiles(0, 0);
    WAIT_ASYNC();
    __syncthreads();

    int buf = 0;
    for (int k0 = 0; k0 < K; k0 += GT_K) {
        if (k0 + GT_K < K) load_tiles(buf ^ 1, k0 + GT_K);   // async prefetch

        v16h afrag[4], bfrag[2];
        #pragma unroll
        for (int mt = 0; mt < 4; mt++)
            afrag[mt] = frag_load(&As[buf][(wm * 64 + mt * 16 + ln) * LDT], koff);
        #pragma unroll
        for (int nt = 0; nt < 2; nt++)
            bfrag[nt] = frag_load(&Bs[buf][(wn * 32 + nt * 16 + ln) * LDT], koff);

        #pragma unroll
        for (int mt = 0; mt < 4; mt++)
            #pragma unroll
            for (int nt = 0; nt < 2; nt++)
                acc[mt][nt] = __builtin_amdgcn_wmma_f32_16x16x32_f16(
                    false, afrag[mt], false, bfrag[nt], (short)0, acc[mt][nt],
                    false, false);

        WAIT_ASYNC();
        __syncthreads();
        buf ^= 1;
    }

    // Epilogue (C/D layout: m = r + 8*half, n = ln per VGPR r).
    #pragma unroll
    for (int mt = 0; mt < 4; mt++) {
        #pragma unroll
        for (int nt = 0; nt < 2; nt++) {
            const int col = colBase + wn * 32 + nt * 16 + ln;
            const float bv = bias[col];
            #pragma unroll
            for (int r = 0; r < 8; r++) {
                const int row = rowBase + wm * 64 + mt * 16 + r + 8 * half;
                float v = acc[mt][nt][r] + bv;
                if (RELU) v = v > 0.0f ? v : 0.0f;
                const size_t off = (size_t)row * N + col;
                if (resid) v += resid[off];
                if (out32) out32[off] = v;
                if (out16) out16[off] = (_Float16)v;
            }
        }
    }
}

// ---------------------------------------------------------------------------
// Fused flash attention, one (b,h) per blockIdx.y, 128 query rows per block.
// 8 waves each own a 16-row strip: S = Q K^T (WMMA), mask, online softmax,
// P (f16, via LDS) @ V (staged transposed) accumulated with WMMA.
// Q/K tiles use the async global->LDS path; V needs a transpose so it keeps
// the VGPR round-trip scatter.
// ---------------------------------------------------------------------------
constexpr int ATT_BM = 128;
constexpr int LQK    = 72;   // 64 + 8 pad f16 => 144B stride (16B aligned)

__global__ void __launch_bounds__(256)
attention_kernel(const _Float16* __restrict__ q, const _Float16* __restrict__ k,
                 const _Float16* __restrict__ v, const unsigned char* __restrict__ mask,
                 _Float16* __restrict__ pooled) {
    __shared__ __align__(16) _Float16 Qs [ATT_BM * LQK];   // [m][d]
    __shared__ __align__(16) _Float16 Ks [64 * LQK];       // [n][d]
    __shared__ __align__(16) _Float16 Vts[64 * LQK];       // [d][n] (transposed)
    __shared__ __align__(16) _Float16 Ps [8 * 16 * LQK];   // per-wave P strip

    const int tid  = threadIdx.x;
    const int wid  = tid >> 5;
    const int lane = tid & 31;
    const int half = lane >> 4;
    const int ln   = lane & 15;
    const int koff = half ? 8 : 0;

    const int m0 = blockIdx.x * ATT_BM;
    const int bh = blockIdx.y;
    const int b  = bh / NH;
    const int h  = bh % NH;

    // Load Q tile (128 x 64 f16) -- async straight into LDS.
    #pragma unroll
    for (int i = 0; i < 4; i++) {
        const int id  = tid + i * 256;       // 0..1023
        const int r   = id >> 3;             // 0..127
        const int seg = (id & 7) << 3;       // 0..56
        copy_b128_to_lds(&Qs[r * LQK + seg],
                         q + ((size_t)(m0 + r) * BATCH + b) * DMODEL + h * DHEAD + seg);
    }

    float Mrow[8], Lrow[8];
    v8f O[4];
    #pragma unroll
    for (int r = 0; r < 8; r++) { Mrow[r] = -1e30f; Lrow[r] = 0.0f; }
    #pragma unroll
    for (int dt = 0; dt < 4; dt++)
        #pragma unroll
        for (int r = 0; r < 8; r++) O[dt][r] = 0.0f;

    for (int n0 = 0; n0 < MSEQ; n0 += 64) {
        // Load K tile [n][d] (async) and V tile transposed [d][n] (manual).
        #pragma unroll
        for (int i = 0; i < 2; i++) {
            const int id  = tid + i * 256;   // 0..511
            const int r   = id >> 3;         // 0..63
            const int seg = (id & 7) << 3;   // 0..56
            const size_t tokoff = ((size_t)(n0 + r) * BATCH + b) * DMODEL + h * DHEAD + seg;
            copy_b128_to_lds(&Ks[r * LQK + seg], k + tokoff);
            union { uint4 u; _Float16 e[8]; } vv;
            vv.u = *reinterpret_cast<const uint4*>(v + tokoff);
            #pragma unroll
            for (int j = 0; j < 8; j++) Vts[(seg + j) * LQK + r] = vv.e[j];
        }
        WAIT_ASYNC();          // covers Q on first iteration as well
        __syncthreads();

        // S strip = Q(16 x 64) @ K^T(64 x 64)   (no 1/sqrt(dh) scale, per ref)
        v8f S[4];
        #pragma unroll
        for (int nt = 0; nt < 4; nt++) {
            v8f s;
            #pragma unroll
            for (int r = 0; r < 8; r++) s[r] = 0.0f;
            #pragma unroll
            for (int ks = 0; ks < 2; ks++) {
                v16h a  = frag_load(&Qs[(wid * 16 + ln) * LQK + ks * 32], koff);
                v16h bb = frag_load(&Ks[(nt  * 16 + ln) * LQK + ks * 32], koff);
                s = __builtin_amdgcn_wmma_f32_16x16x32_f16(false, a, false, bb,
                                                           (short)0, s, false, false);
            }
            const int n = n0 + nt * 16 + ln;
            if (mask[(size_t)n * BATCH + b]) {
                #pragma unroll
                for (int r = 0; r < 8; r++) s[r] = NEG_INF_VAL;
            }
            S[nt] = s;
        }

        // Online softmax per row (reduce across the 16 lanes of each half).
        #pragma unroll
        for (int r = 0; r < 8; r++) {
            float mx = S[0][r];
            #pragma unroll
            for (int nt = 1; nt < 4; nt++) mx = fmaxf(mx, S[nt][r]);
            #pragma unroll
            for (int off = 1; off < 16; off <<= 1) mx = fmaxf(mx, __shfl_xor(mx, off, 16));
            const float mnew  = fmaxf(Mrow[r], mx);
            const float scale = __expf(Mrow[r] - mnew);
            float psum = 0.0f;
            #pragma unroll
            for (int nt = 0; nt < 4; nt++) {
                const float p = __expf(S[nt][r] - mnew);
                S[nt][r] = p;
                psum += p;
            }
            #pragma unroll
            for (int off = 1; off < 16; off <<= 1) psum += __shfl_xor(psum, off, 16);
            Lrow[r] = Lrow[r] * scale + psum;
            Mrow[r] = mnew;
            #pragma unroll
            for (int dt = 0; dt < 4; dt++) O[dt][r] *= scale;
        }

        // Spill P (f16) to this wave's private LDS strip, reload as A-fragment.
        _Float16* pw = &Ps[wid * 16 * LQK];
        #pragma unroll
        for (int nt = 0; nt < 4; nt++)
            #pragma unroll
            for (int r = 0; r < 8; r++)
                pw[(r + 8 * half) * LQK + nt * 16 + ln] = (_Float16)S[nt][r];

        // O += P(16 x 64) @ V(64 x 64)
        #pragma unroll
        for (int ks = 0; ks < 2; ks++) {
            v16h a = frag_load(&pw[ln * LQK + ks * 32], koff);
            #pragma unroll
            for (int dt = 0; dt < 4; dt++) {
                v16h bb = frag_load(&Vts[(dt * 16 + ln) * LQK + ks * 32], koff);
                O[dt] = __builtin_amdgcn_wmma_f32_16x16x32_f16(false, a, false, bb,
                                                               (short)0, O[dt], false, false);
            }
        }
        __syncthreads();
    }

    // Normalize and write pooled (f16) at [(m*B+b)*D + h*64 + d].
    #pragma unroll
    for (int dt = 0; dt < 4; dt++) {
        #pragma unroll
        for (int r = 0; r < 8; r++) {
            const int m = m0 + wid * 16 + r + 8 * half;
            const float denom = Lrow[r];
            const float val = O[dt][r] / (denom > 0.0f ? denom : 1.0f);
            pooled[((size_t)m * BATCH + b) * DMODEL + h * DHEAD + dt * 16 + ln] = (_Float16)val;
        }
    }
}

// ---------------------------------------------------------------------------
// Host-side orchestration.
// Workspace layout (bytes); requires ws_size >= 152 MiB:
//   [  0,  2M) wqT   [ 2M, 4M) wkT   [ 4M, 6M) wvT   [ 6M, 8M) woT
//   [ 8M, 16M) w1T   [16M, 24M) w2T
//   [24M, 56M) x1 (fp32)      [56M, 72M) xn (f16, reused for ln2)
//   [72M, 88M) pooled (f16)
//   [88M,104M) q  [104M,120M) k  [120M,136M) v   -- dead after attention,
//   [88M,152M) h (f16, FFN hidden) reuses the q/k/v region.
// ---------------------------------------------------------------------------
extern "C" void kernel_launch(void* const* d_in, const int* in_sizes, int n_in,
                              void* d_out, int out_size, void* d_ws, size_t ws_size,
                              hipStream_t stream) {
    (void)in_sizes; (void)n_in; (void)out_size; (void)ws_size;

    const float*         x     = (const float*)d_in[0];
    const unsigned char* mask  = (const unsigned char*)d_in[1];
    const float* ln1_g = (const float*)d_in[2];
    const float* ln1_b = (const float*)d_in[3];
    const float* wq    = (const float*)d_in[4];
    const float* bq    = (const float*)d_in[5];
    const float* wk    = (const float*)d_in[6];
    const float* bk    = (const float*)d_in[7];
    const float* wv    = (const float*)d_in[8];
    const float* bv    = (const float*)d_in[9];
    const float* wo    = (const float*)d_in[10];
    const float* bo    = (const float*)d_in[11];
    const float* ln2_g = (const float*)d_in[12];
    const float* ln2_b = (const float*)d_in[13];
    const float* w1    = (const float*)d_in[14];
    const float* b1    = (const float*)d_in[15];
    const float* w2    = (const float*)d_in[16];
    const float* b2    = (const float*)d_in[17];
    float* out = (float*)d_out;

    char* ws = (char*)d_ws;
    const size_t MiB = 1024ull * 1024ull;
    _Float16* wqT    = (_Float16*)(ws +   0 * MiB);
    _Float16* wkT    = (_Float16*)(ws +   2 * MiB);
    _Float16* wvT    = (_Float16*)(ws +   4 * MiB);
    _Float16* woT    = (_Float16*)(ws +   6 * MiB);
    _Float16* w1T    = (_Float16*)(ws +   8 * MiB);
    _Float16* w2T    = (_Float16*)(ws +  16 * MiB);
    float*    x1     = (float*)   (ws +  24 * MiB);
    _Float16* xn     = (_Float16*)(ws +  56 * MiB);
    _Float16* pooled = (_Float16*)(ws +  72 * MiB);
    _Float16* qh     = (_Float16*)(ws +  88 * MiB);
    _Float16* kh     = (_Float16*)(ws + 104 * MiB);
    _Float16* vh     = (_Float16*)(ws + 120 * MiB);
    _Float16* hh     = (_Float16*)(ws +  88 * MiB);   // reuses q/k/v region

    const dim3 tb(32, 8);
    // Weight transposes (+fp32->f16).
    transpose_convert_kernel<<<dim3(DMODEL/32, DMODEL/32), tb, 0, stream>>>(wq, wqT, DMODEL, DMODEL);
    transpose_convert_kernel<<<dim3(DMODEL/32, DMODEL/32), tb, 0, stream>>>(wk, wkT, DMODEL, DMODEL);
    transpose_convert_kernel<<<dim3(DMODEL/32, DMODEL/32), tb, 0, stream>>>(wv, wvT, DMODEL, DMODEL);
    transpose_convert_kernel<<<dim3(DMODEL/32, DMODEL/32), tb, 0, stream>>>(wo, woT, DMODEL, DMODEL);
    transpose_convert_kernel<<<dim3(FFDIM/32,  DMODEL/32), tb, 0, stream>>>(w1, w1T, DMODEL, FFDIM);
    transpose_convert_kernel<<<dim3(DMODEL/32, FFDIM/32),  tb, 0, stream>>>(w2, w2T, FFDIM,  DMODEL);

    // LN1.
    layernorm_kernel<<<ROWS, 256, 0, stream>>>(x, ln1_g, ln1_b, xn);

    // QKV projections (f16 outputs).
    const dim3 gD(ROWS / GT_M, DMODEL / GT_N);
    gemm_f16_kernel<false><<<gD, 256, 0, stream>>>(xn, wqT, bq, nullptr, nullptr, qh, DMODEL, DMODEL);
    gemm_f16_kernel<false><<<gD, 256, 0, stream>>>(xn, wkT, bk, nullptr, nullptr, kh, DMODEL, DMODEL);
    gemm_f16_kernel<false><<<gD, 256, 0, stream>>>(xn, wvT, bv, nullptr, nullptr, vh, DMODEL, DMODEL);

    // Fused attention -> pooled.
    attention_kernel<<<dim3(MSEQ / ATT_BM, BATCH * NH), 256, 0, stream>>>(qh, kh, vh, mask, pooled);

    // Output projection + residual: x1 = x + pooled @ wo + bo.
    gemm_f16_kernel<false><<<gD, 256, 0, stream>>>(pooled, woT, bo, x, x1, nullptr, DMODEL, DMODEL);

    // LN2 (reuse xn).
    layernorm_kernel<<<ROWS, 256, 0, stream>>>(x1, ln2_g, ln2_b, xn);

    // FFN: h = relu(xn @ w1 + b1);  out = h @ w2 + b2 + x1.
    gemm_f16_kernel<true ><<<dim3(ROWS / GT_M, FFDIM / GT_N), 256, 0, stream>>>(
        xn, w1T, b1, nullptr, nullptr, hh, DMODEL, FFDIM);
    gemm_f16_kernel<false><<<gD, 256, 0, stream>>>(hh, w2T, b2, x1, out, nullptr, FFDIM, DMODEL);
}
